// DynamicAggregationRestoration_21577915695509
// MI455X (gfx1250) — compile-verified, matrix operand order checked by reading, not run on hardware
//
#include <hip/hip_runtime.h>
#include <hip/hip_bf16.h>

typedef __attribute__((ext_vector_type(16))) _Float16 v16h;
typedef __attribute__((ext_vector_type(8)))  _Float16 v8h;
typedef __attribute__((ext_vector_type(8)))  float    v8f;

#define HH 128
#define WW 128
#define HW (HH * WW)
#define NB 4
#define CC 128
#define DG 8
#define CG 16
#define IN_CH 384
#define OFF_CH 216
#define K1 (IN_CH * 9)   // 3456
#define K2 (CC * 9)      // 1152
#define KD_PAD 1280      // 8 groups * 160 (144 real + 16 zero pad)

// ---------------------------------------------------------------------------
// Weight pack: fp32 -> f16, w_dcn padded per-group 144 -> 160
// ---------------------------------------------------------------------------
__global__ __launch_bounds__(256) void pack_w_kernel(
    const float* __restrict__ w1, const float* __restrict__ w2,
    const float* __restrict__ woff, const float* __restrict__ wdcn,
    _Float16* __restrict__ w1f, _Float16* __restrict__ w2f,
    _Float16* __restrict__ wofff, _Float16* __restrict__ wdcnf)
{
  int i = blockIdx.x * 256 + threadIdx.x;
  if (i < CC * K1)     w1f[i]   = (_Float16)w1[i];
  if (i < CC * K2)     w2f[i]   = (_Float16)w2[i];
  if (i < OFF_CH * K2) wofff[i] = (_Float16)woff[i];
  if (i < CC * KD_PAD) {
    int o = i / KD_PAD, r = i % KD_PAD;
    int d = r / 160, rr = r % 160;
    _Float16 v = (_Float16)0.f;
    if (rr < 144) v = (_Float16)wdcn[(size_t)o * K2 + d * 144 + rr];
    wdcnf[i] = v;
  }
}

// ---------------------------------------------------------------------------
// Flow-warp + build f16 conv1 input [B,384,H,W] = cat(enc, warped, warp_feat);
// also copy encoder fp32 into out channels [0,128).
// ---------------------------------------------------------------------------
__global__ __launch_bounds__(256) void warp_pack_kernel(
    const float* __restrict__ enc, const float* __restrict__ flow,
    const float* __restrict__ wfeat,
    _Float16* __restrict__ X1, float* __restrict__ out)
{
  int n = blockIdx.x * 256 + threadIdx.x;
  if (n >= NB * HW) return;
  int b = n >> 14, yx = n & (HW - 1);
  int y = yx >> 7, x = yx & (WW - 1);

  float fy = flow[((size_t)b * HW + yx) * 2 + 1];
  float fx = flow[((size_t)b * HW + yx) * 2 + 0];
  float py = (float)y + fy, px = (float)x + fx;
  float y0f = floorf(py), x0f = floorf(px);
  int y0 = (int)y0f, x0 = (int)x0f;
  float wy = py - y0f, wx = px - x0f;
  float w00 = (1.f - wy) * (1.f - wx), w01 = (1.f - wy) * wx;
  float w10 = wy * (1.f - wx),         w11 = wy * wx;
  bool vy0 = (unsigned)y0 < (unsigned)HH, vy1 = (unsigned)(y0 + 1) < (unsigned)HH;
  bool vx0 = (unsigned)x0 < (unsigned)WW, vx1 = (unsigned)(x0 + 1) < (unsigned)WW;

  for (int c = 0; c < CC; ++c) {
    const float* be = enc   + ((size_t)b * CC + c) * HW;
    const float* bw = wfeat + ((size_t)b * CC + c) * HW;
    float ev = be[yx];
    float wv = bw[yx];
    float s = 0.f;
    if (vy0 && vx0) s += w00 * bw[y0 * WW + x0];
    if (vy0 && vx1) s += w01 * bw[y0 * WW + x0 + 1];
    if (vy1 && vx0) s += w10 * bw[(y0 + 1) * WW + x0];
    if (vy1 && vx1) s += w11 * bw[(y0 + 1) * WW + x0 + 1];
    X1[((size_t)b * IN_CH + c) * HW + yx]            = (_Float16)ev;
    X1[((size_t)b * IN_CH + CC + c) * HW + yx]       = (_Float16)s;
    X1[((size_t)b * IN_CH + 2 * CC + c) * HW + yx]   = (_Float16)wv;
    out[((size_t)b * 2 * CC + c) * HW + yx]          = ev;   // encoder passthrough
  }
}

// ---------------------------------------------------------------------------
// Fragment load (ISA 7.12.2, wave32, 16-bit A/B 16x32):
//   lane's row/col index = lane&15; K runs kb..kb+7 and 16+kb..16+kb+7 where
//   kb = (lane>>4)*8.  With K-contiguous LDS rows this is 2x ds_load_b128.
// ---------------------------------------------------------------------------
__device__ __forceinline__ v16h load_frag_row(const _Float16* __restrict__ row,
                                              int lane)
{
  int kb = (lane >> 4) * 8;
  v8h lo = *(const v8h*)(row + kb);
  v8h hi = *(const v8h*)(row + 16 + kb);
  return __builtin_shufflevector(lo, hi, 0, 1, 2, 3, 4, 5, 6, 7,
                                 8, 9, 10, 11, 12, 13, 14, 15);
}

// ---------------------------------------------------------------------------
// Implicit-GEMM 3x3 conv (stride 1, pad 1) with v_wmma_f32_16x16x32_f16.
// Block = 8 waves. Tile: M=128 out-channels x N=64 pixels (one image row half).
// LDS tiles are K-contiguous per row so fragment loads are 2x b128.
// ---------------------------------------------------------------------------
__global__ __launch_bounds__(256) void conv3x3_wmma_kernel(
    const _Float16* __restrict__ X,   // [B, Cin, H, W] f16
    const _Float16* __restrict__ Wp,  // [Cout][Cin*9] f16 row-major
    const float* __restrict__ bias,
    _Float16* __restrict__ Yh,        // f16 out (or null)
    float* __restrict__ Yf,           // f32 out (or null)
    int Cin, int Cout, int relu)
{
  const int Ktot = Cin * 9;
  const int nbase = blockIdx.x * 64;
  const int mbase = blockIdx.y * 128;
  __shared__ __align__(16) _Float16 aT[128 * 32];   // [row][k]   8 KB
  __shared__ __align__(16) _Float16 bT[64 * 32];    // [pixel][k] 4 KB
  const int tid = threadIdx.x, lane = tid & 31, wv = tid >> 5;

  v8f acc[4] = {};
  for (int kc = 0; kc < Ktot; kc += 32) {
    // --- A tile: 128 rows x 32 k; per thread one 16-half (32B) chunk ---
    {
      int row = tid >> 1, kof = (tid & 1) * 16;
      int gr = mbase + row;
      v8h* dst = (v8h*)(aT + row * 32 + kof);
      if (gr < Cout) {
        const v8h* src = (const v8h*)(Wp + (size_t)gr * Ktot + kc + kof);
        dst[0] = src[0];
        dst[1] = src[1];
        if (kc + 32 < Ktot) __builtin_prefetch(src + 4, 0, 0);
      } else {
        v8h z = (v8h)(_Float16)0.f;
        dst[0] = z;
        dst[1] = z;
      }
    }
    // --- B tile (im2col, [pixel][k]): thread owns one k, 8 contiguous px ---
    {
      int kk = tid >> 3;            // 0..31
      int p0 = (tid & 7) * 8;       // 0,8,..,56
      int k = kc + kk;
      int cin = k / 9, tap = k % 9;
      int dy = tap / 3 - 1, dx = tap % 3 - 1;
      int n0 = nbase + p0;
      int b = n0 >> 14, yx = n0 & (HW - 1);
      int y = yx >> 7, x0 = yx & (WW - 1);
      int yy = y + dy;
      const _Float16* src = X + ((size_t)b * Cin + cin) * HW + yy * WW;
      bool rowok = (unsigned)yy < (unsigned)HH;
#pragma unroll
      for (int e = 0; e < 8; ++e) {
        int xx = x0 + e + dx;
        _Float16 v = (_Float16)0.f;
        if (rowok && (unsigned)xx < (unsigned)WW) v = src[xx];
        bT[(p0 + e) * 32 + kk] = v;
      }
    }
    __syncthreads();

    v16h a = load_frag_row(aT + (wv * 16 + (lane & 15)) * 32, lane);
#pragma unroll
    for (int j = 0; j < 4; ++j) {
      v16h bf = load_frag_row(bT + (j * 16 + (lane & 15)) * 32, lane);
      acc[j] = __builtin_amdgcn_wmma_f32_16x16x32_f16(
          false, a, false, bf, (short)0, acc[j], false, false);
    }
    __syncthreads();
  }

  // --- epilogue: bias + lrelu, write f16 and/or f32 ---
#pragma unroll
  for (int j = 0; j < 4; ++j) {
#pragma unroll
    for (int r = 0; r < 8; ++r) {
      int m = mbase + wv * 16 + r + 8 * (lane >> 4);
      int p = nbase + j * 16 + (lane & 15);
      if (m < Cout) {
        float v = acc[j][r] + bias[m];
        if (relu) v = (v >= 0.f) ? v : 0.1f * v;
        int b = p >> 14, yx = p & (HW - 1);
        size_t o = ((size_t)b * Cout + m) * HW + yx;
        if (Yh) Yh[o] = (_Float16)v;
        if (Yf) Yf[o] = v;
      }
    }
  }
}

// ---------------------------------------------------------------------------
// Modulated deformable conv as WMMA GEMM: M=128, K=8*160 (padded), N=64 px.
// Per group: compute offsets/masks, bilinear-sample modulated columns into
// LDS [64 px][160 k] f16 (K-contiguous), then 5 WMMA K-steps.
// ---------------------------------------------------------------------------
__global__ __launch_bounds__(256) void dcn_wmma_kernel(
    const float* __restrict__ wfeat,      // [B,128,H,W]
    const float* __restrict__ om,         // [B,216,H,W]
    const float* __restrict__ pre_off,    // [B,9,H,W,2]
    const float* __restrict__ pre_sim,    // [B,1,H,W]
    const _Float16* __restrict__ Wd,      // [128][1280] f16 padded
    const float* __restrict__ bias,
    float* __restrict__ out)              // [B,256,H,W], channels 128..255
{
  const int nbase = blockIdx.x * 64;
  __shared__ __align__(16) _Float16 colsT[64 * 160];  // [pixel][k] 20 KB
  __shared__ __align__(16) _Float16 aT[128 * 32];     // [row][k]    8 KB
  __shared__ float sy[576], sx[576], sm[576];         // 64 px * 9 taps
  const int tid = threadIdx.x, lane = tid & 31, wv = tid >> 5;

  v8f acc[4] = {};
  for (int d = 0; d < DG; ++d) {
    // --- per-(pixel,tap) sample coords + modulation mask ---
    for (int i = tid; i < 576; i += 256) {
      int p = i / 9, k = i % 9;
      int n = nbase + p;
      int b = n >> 14, yx = n & (HW - 1);
      int y = yx >> 7, x = yx & (WW - 1);
      int j = d * 9 + k;
      float oy = om[((size_t)b * OFF_CH + 2 * j) * HW + yx]
               + pre_off[(((size_t)b * 9 + k) * HW + yx) * 2 + 1];
      float ox = om[((size_t)b * OFF_CH + 2 * j + 1) * HW + yx]
               + pre_off[(((size_t)b * 9 + k) * HW + yx) * 2 + 0];
      float mr = om[((size_t)b * OFF_CH + 144 + j) * HW + yx];
      float msk = (1.f / (1.f + expf(-mr))) * pre_sim[(size_t)b * HW + yx];
      sy[i] = (float)y + (float)(k / 3) - 1.f + oy;
      sx[i] = (float)x + (float)(k % 3) - 1.f + ox;
      sm[i] = msk;
    }
    // zero K-pad columns 144..159 of each pixel row
    for (int i = tid; i < 64 * 16; i += 256) {
      int p = i >> 4, kr = 144 + (i & 15);
      colsT[p * 160 + kr] = (_Float16)0.f;
    }
    __syncthreads();

    // --- build modulated columns: fixed pixel per thread, kr strided by 4 ---
    for (int i = tid; i < 144 * 64; i += 256) {
      int kr = i >> 6, p = i & 63;
      int c = kr / 9, k = kr % 9;
      int n = nbase + p;
      int b = n >> 14;
      int e = p * 9 + k;
      float py = sy[e], px = sx[e], msk = sm[e];
      float y0f = floorf(py), x0f = floorf(px);
      int y0 = (int)y0f, x0 = (int)x0f;
      float wy = py - y0f, wx = px - x0f;
      const float* base = wfeat + ((size_t)b * CC + d * CG + c) * HW;
      float v = 0.f;
      if ((unsigned)y0 < (unsigned)HH) {
        if ((unsigned)x0 < (unsigned)WW)       v += (1.f - wy) * (1.f - wx) * base[y0 * WW + x0];
        if ((unsigned)(x0 + 1) < (unsigned)WW) v += (1.f - wy) * wx         * base[y0 * WW + x0 + 1];
      }
      if ((unsigned)(y0 + 1) < (unsigned)HH) {
        if ((unsigned)x0 < (unsigned)WW)       v += wy * (1.f - wx) * base[(y0 + 1) * WW + x0];
        if ((unsigned)(x0 + 1) < (unsigned)WW) v += wy * wx         * base[(y0 + 1) * WW + x0 + 1];
      }
      colsT[p * 160 + kr] = (_Float16)(v * msk);
    }
    __syncthreads();

    // --- K loop over this group's padded 160 (5 chunks of 32) ---
    for (int kc2 = 0; kc2 < 5; ++kc2) {
      {
        int row = tid >> 1, kof = (tid & 1) * 16;
        const v8h* src = (const v8h*)(Wd + (size_t)row * KD_PAD + d * 160 + kc2 * 32 + kof);
        v8h* dst = (v8h*)(aT + row * 32 + kof);
        dst[0] = src[0];
        dst[1] = src[1];
      }
      __syncthreads();
      v16h a = load_frag_row(aT + (wv * 16 + (lane & 15)) * 32, lane);
#pragma unroll
      for (int j = 0; j < 4; ++j) {
        v16h bf = load_frag_row(colsT + (j * 16 + (lane & 15)) * 160 + kc2 * 32, lane);
        acc[j] = __builtin_amdgcn_wmma_f32_16x16x32_f16(
            false, a, false, bf, (short)0, acc[j], false, false);
      }
      __syncthreads();
    }
  }

  // --- epilogue: bias + lrelu -> out channels [128,256) ---
#pragma unroll
  for (int j = 0; j < 4; ++j) {
#pragma unroll
    for (int r = 0; r < 8; ++r) {
      int m = wv * 16 + r + 8 * (lane >> 4);
      int p = nbase + j * 16 + (lane & 15);
      float v = acc[j][r] + bias[m];
      v = (v >= 0.f) ? v : 0.1f * v;
      int b = p >> 14, yx = p & (HW - 1);
      out[((size_t)b * 2 * CC + CC + m) * HW + yx] = v;
    }
  }
}

// ---------------------------------------------------------------------------
extern "C" void kernel_launch(void* const* d_in, const int* in_sizes, int n_in,
                              void* d_out, int out_size, void* d_ws, size_t ws_size,
                              hipStream_t stream) {
  const float* pre_offset = (const float*)d_in[0];
  const float* pre_sim    = (const float*)d_in[1];
  const float* enc        = (const float*)d_in[2];
  const float* flow       = (const float*)d_in[3];
  const float* wfeat      = (const float*)d_in[4];
  const float* w1   = (const float*)d_in[5];
  const float* b1   = (const float*)d_in[6];
  const float* w2   = (const float*)d_in[7];
  const float* b2   = (const float*)d_in[8];
  const float* woff = (const float*)d_in[9];
  const float* boff = (const float*)d_in[10];
  const float* wdcn = (const float*)d_in[11];
  const float* bdcn = (const float*)d_in[12];
  float* out = (float*)d_out;

  char* ws = (char*)d_ws;
  size_t off = 0;
  auto carve = [&](size_t bytes) -> void* {
    void* p = ws + off;
    off += (bytes + 255) & ~(size_t)255;
    return p;
  };
  _Float16* X1    = (_Float16*)carve((size_t)NB * IN_CH * HW * 2);   // 50.3 MB
  _Float16* h1    = (_Float16*)carve((size_t)NB * CC * HW * 2);      // 16.8 MB
  _Float16* h2    = (_Float16*)carve((size_t)NB * CC * HW * 2);      // 16.8 MB
  float*    omBuf = (float*)   carve((size_t)NB * OFF_CH * HW * 4);  // 56.6 MB
  _Float16* w1f   = (_Float16*)carve((size_t)CC * K1 * 2);
  _Float16* w2f   = (_Float16*)carve((size_t)CC * K2 * 2);
  _Float16* wofff = (_Float16*)carve((size_t)OFF_CH * K2 * 2);
  _Float16* wdcnf = (_Float16*)carve((size_t)CC * KD_PAD * 2);

  // 1) pack weights to f16 (largest table: 128*3456 = 442368 elements)
  pack_w_kernel<<<(CC * K1 + 255) / 256, 256, 0, stream>>>(
      w1, w2, woff, wdcn, w1f, w2f, wofff, wdcnf);

  // 2) flow-warp + conv1 input assembly + encoder passthrough
  warp_pack_kernel<<<(NB * HW + 255) / 256, 256, 0, stream>>>(
      enc, flow, wfeat, X1, out);

  const int ntiles = NB * HW / 64;  // 1024
  // 3) conv1: 384 -> 128, lrelu, f16 out
  conv3x3_wmma_kernel<<<dim3(ntiles, 1), 256, 0, stream>>>(
      X1, w1f, b1, h1, nullptr, IN_CH, CC, 1);
  // 4) conv2: 128 -> 128, lrelu, f16 out
  conv3x3_wmma_kernel<<<dim3(ntiles, 1), 256, 0, stream>>>(
      h1, w2f, b2, h2, nullptr, CC, CC, 1);
  // 5) conv_offset: 128 -> 216, no act, f32 out
  conv3x3_wmma_kernel<<<dim3(ntiles, 2), 256, 0, stream>>>(
      h2, wofff, boff, nullptr, omBuf, CC, OFF_CH, 0);

  // 6) modulated deformable conv + lrelu -> out channels [128,256)
  dcn_wmma_kernel<<<ntiles, 256, 0, stream>>>(
      wfeat, omBuf, pre_offset, pre_sim, wdcnf, bdcn, out);
}
